// GCN4Rec_15023795602160
// MI455X (gfx1250) — matrix operand order repaired and checked by reading, not compiled, not run on hardware
//
#include <hip/hip_runtime.h>
#include <hip/hip_bf16.h>
#include <math.h>

typedef __attribute__((ext_vector_type(16))) _Float16 v16h;
typedef __attribute__((ext_vector_type(8)))  _Float16 v8h;
typedef __attribute__((ext_vector_type(2)))  _Float16 v2h;
typedef __attribute__((ext_vector_type(8)))  float    v8f;

#define DIM 64
#define HID 128

__device__ __forceinline__ float wave_reduce_sum(float v) {
#pragma unroll
  for (int m = 16; m >= 1; m >>= 1)
    v += __shfl_xor(v, m, 32);
  return v;
}

// ---------------------------------------------------------------------------
// 1. Renorm rows (max_norm = 1) and convert to f16.  One wave32 per row of 64.
// ---------------------------------------------------------------------------
__global__ void k_renorm_f16(const float* __restrict__ emb,
                             _Float16* __restrict__ out, int n) {
  unsigned wave = (blockIdx.x * blockDim.x + threadIdx.x) >> 5;
  unsigned lane = threadIdx.x & 31u;
  if (wave >= (unsigned)n) return;
  const float* row = emb + (size_t)wave * DIM;
  float a = row[lane * 2 + 0];
  float b = row[lane * 2 + 1];
  float ss = wave_reduce_sum(a * a + b * b);
  float s = fminf(1.0f, 1.0f / fmaxf(sqrtf(ss), 1e-12f));
  v2h o;
  o[0] = (_Float16)(a * s);
  o[1] = (_Float16)(b * s);
  *(v2h*)(out + (size_t)wave * DIM + lane * 2) = o;
}

// ---------------------------------------------------------------------------
// 2. Degree accumulation + finalize.
// ---------------------------------------------------------------------------
__global__ void k_deg_acc(const int* __restrict__ dst, float* __restrict__ deg, int E) {
  unsigned e = blockIdx.x * blockDim.x + threadIdx.x;
  if (e < (unsigned)E) atomicAdd(&deg[dst[e]], 1.0f);
}

__global__ void k_deg_fin(const float* __restrict__ deg, float* __restrict__ dinv,
                          float* __restrict__ invdeg, int n) {
  unsigned idx = blockIdx.x * blockDim.x + threadIdx.x;
  if (idx < (unsigned)n) {
    float d = deg[idx] + 1.0f;          // self loop
    dinv[idx] = rsqrtf(d);
    invdeg[idx] = 1.0f / d;
  }
}

// ---------------------------------------------------------------------------
// 3a. Pre-pack W (f32, K x NCOLS row-major) into WMMA B-fragment layout (f16):
//     Bfrag[((nt*KT + kt)*32 + lane)*16 + j] = W[(kt*32 + (lane>>4)*16 + j)*NCOLS
//                                                + nt*16 + (lane&15)]
//     One wave per (coltile, ktile) fragment.  Runs once per launch; tiny.
// ---------------------------------------------------------------------------
template <int K, int NCOLS>
__global__ void k_prep_bfrag(const float* __restrict__ W, _Float16* __restrict__ Bfrag) {
  constexpr int KT = K / 32;
  constexpr int NT = NCOLS / 16;
  unsigned wave = (blockIdx.x * blockDim.x + threadIdx.x) >> 5;
  unsigned lane = threadIdx.x & 31u;
  if (wave >= (unsigned)(NT * KT)) return;
  int nt = (int)(wave / KT);
  int kt = (int)(wave - nt * KT);
  int ml = (int)(lane & 15u);
  int hi = (int)(lane >> 4);
  int col = nt * 16 + ml;
  v16h b;
#pragma unroll
  for (int j = 0; j < 16; j++)
    b[j] = (_Float16)W[(size_t)(kt * 32 + hi * 16 + j) * NCOLS + col];
  *(v16h*)(Bfrag + ((size_t)(nt * KT + kt) * 32 + lane) * 16) = b;
}

// ---------------------------------------------------------------------------
// 3b. WMMA GEMM: C[n x NCOLS] = A_f16[n x K] * W[K x NCOLS] (pre-packed frags).
//     One wave per 16x16 output tile; inner loop is 2 vector loads + 1 vector
//     load + v_wmma_f32_16x16x32_f16.
//     A fragment (contiguous k, matching Bfrag): lane<16 -> K base 0, lane>=16 -> +16;
//     each lane: 16 consecutive halfs at kbase.
// ---------------------------------------------------------------------------
template <int K, int NCOLS>
__global__ void k_gemm_wmma(const _Float16* __restrict__ A,
                            const _Float16* __restrict__ Bfrag,
                            float* __restrict__ C, int n) {
  constexpr int KT = K / 32;
  unsigned wave = (blockIdx.x * blockDim.x + threadIdx.x) >> 5;
  unsigned lane = threadIdx.x & 31u;
  const int tilesPerRow = NCOLS / 16;
  unsigned mt = wave / tilesPerRow;
  unsigned nt = wave - mt * tilesPerRow;
  int row0 = (int)mt * 16;
  if (row0 >= n) return;               // wave-uniform exit (EXEC stays all-1)
  int ml = (int)(lane & 15u);
  int hi = (int)(lane >> 4);
  int mrow = row0 + ml;
  if (mrow > n - 1) mrow = n - 1;      // clamp loads for partial tail tile

  const _Float16* bp = Bfrag + ((size_t)nt * KT * 32 + lane) * 16;
  v8f acc = {};
#pragma unroll
  for (int kt = 0; kt < KT; kt++) {
    const _Float16* ap = A + (size_t)mrow * K + kt * 32 + hi * 16;
    v8h alo = *(const v8h*)ap;
    v8h ahi = *(const v8h*)(ap + 8);
    v16h a;
#pragma unroll
    for (int j = 0; j < 8; j++) { a[j] = alo[j]; a[j + 8] = ahi[j]; }

    v16h b = *(const v16h*)(bp + (size_t)kt * 32 * 16);

    acc = __builtin_amdgcn_wmma_f32_16x16x32_f16(
        /*neg_a=*/false, a, /*neg_b=*/false, b,
        /*c_mod=*/(short)0, acc, /*reuse_a=*/false, /*reuse_b=*/false);
  }

  int col = (int)nt * 16 + ml;
  float* cp = C + (size_t)(row0 + hi * 8) * NCOLS + col;
  if (row0 + 16 <= n) {
    // full tile: straight-line stores (64B-coalesced across lanes per row)
#pragma unroll
    for (int r = 0; r < 8; r++) cp[(size_t)r * NCOLS] = acc[r];
  } else {
    int rbase = row0 + hi * 8;
#pragma unroll
    for (int r = 0; r < 8; r++)
      if (rbase + r < n) cp[(size_t)r * NCOLS] = acc[r];
  }
}

// ---------------------------------------------------------------------------
// 4. Edge scatter: out[dst] += h[src] * dinv[src]*dinv[dst].  Wave per edge.
//    h1/out1 (51 MB each) are L2-resident on MI455X (192 MB L2).
// ---------------------------------------------------------------------------
__global__ void k_scatter128(const float* __restrict__ H, float* __restrict__ O,
                             const int* __restrict__ src, const int* __restrict__ dst,
                             const float* __restrict__ dinv, int E) {
  unsigned wave = (blockIdx.x * blockDim.x + threadIdx.x) >> 5;
  unsigned lane = threadIdx.x & 31u;
  if (wave >= (unsigned)E) return;
  int s = src[wave], d = dst[wave];
  float nrm = dinv[s] * dinv[d];
  const float4 h = *(const float4*)(H + (size_t)s * HID + lane * 4);
  float* op = O + (size_t)d * HID + lane * 4;
  atomicAdd(op + 0, h.x * nrm);
  atomicAdd(op + 1, h.y * nrm);
  atomicAdd(op + 2, h.z * nrm);
  atomicAdd(op + 3, h.w * nrm);
}

__global__ void k_scatter64(const float* __restrict__ H, float* __restrict__ O,
                            const int* __restrict__ src, const int* __restrict__ dst,
                            const float* __restrict__ dinv, int E) {
  unsigned wave = (blockIdx.x * blockDim.x + threadIdx.x) >> 5;
  unsigned lane = threadIdx.x & 31u;
  if (wave >= (unsigned)E) return;
  int s = src[wave], d = dst[wave];
  float nrm = dinv[s] * dinv[d];
  const float2 h = *(const float2*)(H + (size_t)s * DIM + lane * 2);
  float* op = O + (size_t)d * DIM + lane * 2;
  atomicAdd(op + 0, h.x * nrm);
  atomicAdd(op + 1, h.y * nrm);
}

// ---------------------------------------------------------------------------
// 5. Self-loop term + bias + ReLU, convert to f16 for the next GEMM.
// ---------------------------------------------------------------------------
__global__ void k_self_relu_f16(const float* __restrict__ O, const float* __restrict__ H,
                                const float* __restrict__ invdeg, const float* __restrict__ bias,
                                _Float16* __restrict__ X, int total) {
  unsigned e = blockIdx.x * blockDim.x + threadIdx.x;
  if (e >= (unsigned)total) return;
  unsigned rowi = e >> 7;   // /HID
  unsigned j = e & 127u;
  float v = O[e] + H[e] * invdeg[rowi] + bias[j];
  X[e] = (_Float16)fmaxf(v, 0.0f);
}

// ---------------------------------------------------------------------------
// 6. Final: gather item row (finishing conv2 self term on the fly), renorm
//    user row, dot, sigmoid.  Wave per sample.
// ---------------------------------------------------------------------------
__global__ void k_final(const int* __restrict__ u, const int* __restrict__ ii,
                        const float* __restrict__ user_emb,
                        const float* __restrict__ O2, const float* __restrict__ H2,
                        const float* __restrict__ invdeg, const float* __restrict__ b2,
                        float* __restrict__ out, int B) {
  unsigned wave = (blockIdx.x * blockDim.x + threadIdx.x) >> 5;
  unsigned lane = threadIdx.x & 31u;
  if (wave >= (unsigned)B) return;
  int iu = u[wave], it = ii[wave];
  int j0 = (int)lane * 2, j1 = j0 + 1;
  float id = invdeg[it];
  const float* o2 = O2 + (size_t)it * DIM;
  const float* h2 = H2 + (size_t)it * DIM;
  float i0 = o2[j0] + h2[j0] * id + b2[j0];
  float i1 = o2[j1] + h2[j1] * id + b2[j1];
  float u0 = user_emb[(size_t)iu * DIM + j0];
  float u1 = user_emb[(size_t)iu * DIM + j1];
  float ss = wave_reduce_sum(u0 * u0 + u1 * u1);
  float s = fminf(1.0f, 1.0f / fmaxf(sqrtf(ss), 1e-12f));
  float dot = wave_reduce_sum(u0 * i0 + u1 * i1) * s;
  if (lane == 0) out[wave] = 1.0f / (1.0f + expf(-dot));
}

// ---------------------------------------------------------------------------
extern "C" void kernel_launch(void* const* d_in, const int* in_sizes, int n_in,
                              void* d_out, int out_size, void* d_ws, size_t ws_size,
                              hipStream_t stream) {
  const int*   u          = (const int*)d_in[0];
  const int*   iidx       = (const int*)d_in[1];
  const int*   edges      = (const int*)d_in[2];
  const float* user_emb   = (const float*)d_in[3];
  const float* entity_emb = (const float*)d_in[4];
  const float* W1         = (const float*)d_in[5];
  const float* b1         = (const float*)d_in[6];
  const float* W2         = (const float*)d_in[7];
  const float* b2         = (const float*)d_in[8];

  const int Bn = in_sizes[0];
  const int E  = in_sizes[2] / 2;
  const int N  = in_sizes[4] / DIM;
  const int* src = edges;
  const int* dst = edges + E;

  // Workspace layout (aliased: h2 reuses h1 space, out2 reuses out1 space).
  char* ws = (char*)d_ws;
  size_t off = 0;
  auto take = [&](size_t bytes) -> char* {
    char* p = ws + off;
    off = (off + bytes + 255) & ~(size_t)255;
    return p;
  };
  float*    deg    = (float*)take((size_t)N * 4);
  float*    dinv   = (float*)take((size_t)N * 4);
  float*    invdeg = (float*)take((size_t)N * 4);
  _Float16* xh     = (_Float16*)take((size_t)N * DIM * 2);
  _Float16* x1h    = (_Float16*)take((size_t)N * HID * 2);
  _Float16* w1frag = (_Float16*)take((size_t)DIM * HID * 2);   // 16 KB
  _Float16* w2frag = (_Float16*)take((size_t)HID * DIM * 2);   // 16 KB
  float*    hbuf   = (float*)take((size_t)N * HID * 4);        // h1, then h2 (N*DIM)
  float*    obuf   = (float*)take((size_t)N * HID * 4);        // out1, then out2 (N*DIM)
  (void)ws_size; (void)n_in; (void)out_size;

  const int TPB = 256;
  auto blocks = [](size_t threads, int tpb) -> unsigned {
    return (unsigned)((threads + tpb - 1) / tpb);
  };

  // 0. pre-pack W1/W2 into WMMA B-fragment layout (16 waves each)
  k_prep_bfrag<DIM, HID><<<2, TPB, 0, stream>>>(W1, w1frag);
  k_prep_bfrag<HID, DIM><<<2, TPB, 0, stream>>>(W2, w2frag);

  // 1. renorm entities -> f16
  k_renorm_f16<<<blocks((size_t)N * 32, TPB), TPB, 0, stream>>>(entity_emb, xh, N);

  // 2. degrees
  hipMemsetAsync(deg, 0, (size_t)N * 4, stream);
  k_deg_acc<<<blocks((size_t)E, TPB), TPB, 0, stream>>>(dst, deg, E);
  k_deg_fin<<<blocks((size_t)N, TPB), TPB, 0, stream>>>(deg, dinv, invdeg, N);

  // 3. GEMM1: h1 = x @ W1   [N x 64] x [64 x 128]
  {
    size_t waves = (size_t)((N + 15) / 16) * (HID / 16);
    k_gemm_wmma<DIM, HID><<<blocks(waves * 32, TPB), TPB, 0, stream>>>(xh, w1frag, hbuf, N);
  }

  // 4. scatter conv1
  hipMemsetAsync(obuf, 0, (size_t)N * HID * 4, stream);
  k_scatter128<<<blocks((size_t)E * 32, TPB), TPB, 0, stream>>>(hbuf, obuf, src, dst, dinv, E);

  // 5. self term + bias + relu -> f16 x1
  k_self_relu_f16<<<blocks((size_t)N * HID, TPB), TPB, 0, stream>>>(obuf, hbuf, invdeg, b1,
                                                                    x1h, N * HID);

  // 6. GEMM2: h2 = x1 @ W2  [N x 128] x [128 x 64]   (h2 overwrites hbuf)
  {
    size_t waves = (size_t)((N + 15) / 16) * (DIM / 16);
    k_gemm_wmma<HID, DIM><<<blocks(waves * 32, TPB), TPB, 0, stream>>>(x1h, w2frag, hbuf, N);
  }

  // 7. scatter conv2 (out2 overwrites obuf)
  hipMemsetAsync(obuf, 0, (size_t)N * DIM * 4, stream);
  k_scatter64<<<blocks((size_t)E * 32, TPB), TPB, 0, stream>>>(hbuf, obuf, src, dst, dinv, E);

  // 8. final gather + user renorm + dot + sigmoid
  k_final<<<blocks((size_t)Bn * 32, TPB), TPB, 0, stream>>>(u, iidx, user_emb, obuf, hbuf,
                                                            invdeg, b2, (float*)d_out, Bn);
}
